// BitcoinLSTM_22402549416725
// MI455X (gfx1250) — matrix operand order repaired and checked by reading, not compiled, not run on hardware
//
#include <hip/hip_runtime.h>
#include <hip/hip_bf16.h>
#include <math.h>

// Problem constants (from reference): B=256, T=256, I=16, H=512, L=4
#define BB 256
#define TT 256
#define II 16
#define HH 512
#define GG 2048              // 4*H gate columns
#define NWG 128              // 4 batch tiles (64 rows) x 32 hidden tiles (16 cols)
#define SCAN_THREADS 128     // 4 waves of 32

typedef __attribute__((ext_vector_type(16))) __bf16 v16bf;
typedef __attribute__((ext_vector_type(8)))  __bf16 v8bf;
typedef __attribute__((ext_vector_type(8)))  float  v8f;
typedef int v4i_vs __attribute__((vector_size(4 * sizeof(int))));

#define AS1 __attribute__((address_space(1)))
#define AS3 __attribute__((address_space(3)))

// gfx1250 async global->LDS copy path (ASYNCcnt-tracked, no VGPR round-trip).
// Signature (from hipcc diagnostic): (AS1 v4i*, AS3 v4i*, imm offset, imm cpol)
#if __has_builtin(__builtin_amdgcn_global_load_async_to_lds_b128)
#define HAVE_ASYNC_LDS 1
#else
#define HAVE_ASYNC_LDS 0
#endif

__device__ __forceinline__ void g2l_16B(const void* g, void* l) {
#if HAVE_ASYNC_LDS
    __builtin_amdgcn_global_load_async_to_lds_b128(
        (AS1 v4i_vs*)g, (AS3 v4i_vs*)l, 0, 0);
#else
    *(v8bf*)l = *(const v8bf*)g;
#endif
}

__device__ __forceinline__ void wait_async_done() {
#if HAVE_ASYNC_LDS
#if __has_builtin(__builtin_amdgcn_s_wait_asynccnt)
    __builtin_amdgcn_s_wait_asynccnt(0);
#else
    asm volatile("s_wait_asynccnt 0" ::: "memory");
#endif
#endif
}

__device__ __forceinline__ float sigm(float x) { return 1.0f / (1.0f + __expf(-x)); }

// ---------------------------------------------------------------------------
// Prep: x (f32 [B,T,16]) -> bf16 [B,T,32] (K padded to 32 for WMMA K-steps)
// ---------------------------------------------------------------------------
__global__ void prep_x_kernel(const float* __restrict__ x, __bf16* __restrict__ xb) {
    const size_t total = (size_t)BB * TT * 32;
    for (size_t i = (size_t)blockIdx.x * blockDim.x + threadIdx.x; i < total;
         i += (size_t)gridDim.x * blockDim.x) {
        int k = (int)(i & 31);
        size_t bt = i >> 5;
        xb[i] = (k < II) ? (__bf16)x[bt * II + k] : (__bf16)0.0f;
    }
}

// ---------------------------------------------------------------------------
// Prep: combined bias = b_ih + b_hh (f32, 2048)
// ---------------------------------------------------------------------------
__global__ void prep_bias_kernel(const float* __restrict__ bi, const float* __restrict__ bh,
                                 float* __restrict__ out) {
    int i = blockIdx.x * blockDim.x + threadIdx.x;
    if (i < GG) out[i] = bi[i] + bh[i];
}

// ---------------------------------------------------------------------------
// Prep: pre-swizzle [W_ih ; W_hh] (row n = gate col, col k) into bf16 WMMA
// B-fragment order. Flat layout: ((ntile*nKK + kk)*32 + lane)*16 + e
// where n = ntile*16 + (lane&15), k = kk*32 + (lane>>4)*16 + e  (contiguous k order, same as the A operand).
// k < Kxp -> W_ih (zero-padded past Kx), else W_hh[k - Kxp].
// ---------------------------------------------------------------------------
__global__ void prep_wfrag_kernel(const float* __restrict__ w_ih, const float* __restrict__ w_hh,
                                  int Kx, int Kxp, __bf16* __restrict__ wfrag) {
    const int Ktot = Kxp + HH;
    const int nKK  = Ktot / 32;
    const size_t total = (size_t)GG * Ktot;
    for (size_t idx = (size_t)blockIdx.x * blockDim.x + threadIdx.x; idx < total;
         idx += (size_t)gridDim.x * blockDim.x) {
        int e    = (int)(idx & 15);
        int lane = (int)((idx >> 4) & 31);
        size_t rem = idx >> 9;
        int kk    = (int)(rem % nKK);
        int ntile = (int)(rem / nKK);
        int n = ntile * 16 + (lane & 15);
        int k = kk * 32 + ((lane >> 4) << 4) + e;
        float v;
        if (k < Kxp) v = (k < Kx) ? w_ih[(size_t)n * Kx + k] : 0.0f;
        else         v = w_hh[(size_t)n * HH + (k - Kxp)];
        wfrag[idx] = (__bf16)v;
    }
}

// ---------------------------------------------------------------------------
// Persistent LSTM scan for one layer.
//   grid = 128 WGs (4 batch tiles x 64 rows  x  32 hidden tiles x 16 cols),
//   block = 128 (4 waves). Weight fragments pinned in LDS for all T steps.
//   Per step: async-stage x_t | h_{t-1} to LDS, K-loop of
//   v_wmma_f32_16x16x32_bf16 (4 gates), f32 activations + cell state in regs,
//   device-scope barrier between steps; h-state ping-pong across steps.
// ---------------------------------------------------------------------------
__device__ __forceinline__ void grid_barrier(unsigned* cnt, unsigned target) {
    __syncthreads();
    if (threadIdx.x == 0) {
        __threadfence();
        atomicAdd(cnt, 1u);
        while (__hip_atomic_load(cnt, __ATOMIC_ACQUIRE, __HIP_MEMORY_SCOPE_AGENT) < target)
            __builtin_amdgcn_s_sleep(2);
    }
    __syncthreads();
}

__global__ void lstm_scan_kernel(const __bf16* __restrict__ xin,   // [B,T,Kxp] bf16
                                 int Kxp,
                                 const __bf16* __restrict__ wfrag, // fragment-ordered weights
                                 const float*  __restrict__ bias,  // [2048] f32
                                 __bf16* __restrict__ hstA,        // [B,H] ping
                                 __bf16* __restrict__ hstB,        // [B,H] pong
                                 __bf16* __restrict__ hout,        // [B,T,H] layer output
                                 unsigned* __restrict__ counter) {
    extern __shared__ __align__(32) char smem_raw[];
    const int Ktot = Kxp + HH;
    const int nKK  = Ktot / 32;
    const int nKKx = Kxp / 32;
    const int tile_halves = Ktot * 16;                 // per gate-tile fragment block

    __bf16* wlds  = (__bf16*)smem_raw;                 // 4 * tile_halves
    __bf16* stage = wlds + 4 * tile_halves;            // 64 rows x Ktot (x | h)

    const int wg    = blockIdx.x;
    const int btile = wg >> 5;                         // 0..3
    const int hi    = wg & 31;                         // 0..31 hidden tile
    const int tid   = threadIdx.x;
    const int lane  = tid & 31;
    const int wave  = tid >> 5;                        // 0..3
    const int rbase = wave * 16;                       // rows within 64-row tile
    const int rowg0 = btile * 64;                      // global batch row base

    // ---- Pin weight fragments for the 4 gates in LDS (reused for all T steps)
    for (int g = 0; g < 4; ++g) {
        const int ntile = g * 32 + hi;
        const __bf16* src = wfrag + (size_t)ntile * tile_halves;
        __bf16* dst = wlds + g * tile_halves;
        for (int i = tid * 8; i < tile_halves; i += SCAN_THREADS * 8)
            g2l_16B(src + i, dst + i);
    }

    // Bias per gate column (constant across M rows and accumulator regs)
    float bv[4];
#pragma unroll
    for (int g = 0; g < 4; ++g) bv[g] = bias[g * HH + hi * 16 + (lane & 15)];

    v8f c = {};                                        // cell state, f32, in registers
    wait_async_done();
    __syncthreads();

    // Per-lane A-fragment addressing (16-bit A 16x32, contiguous k = 16*(lane>>4) + e, same as B)
    const int arow = rbase + (lane & 15);
    const int koff = (lane >> 4) << 4;

    for (int t = 0; t < TT; ++t) {
        // ---- stage h_{t-1} tile first (skip t=0; h0 == 0), then x_t tile;
        //      both via ASYNCcnt-tracked global->LDS copies.
        const __bf16* hread  = (t & 1) ? hstB : hstA;
        __bf16*       hwrite = (t & 1) ? hstA : hstB;
        if (t > 0) {
            const int cpr = HH / 8;
            for (int i = tid; i < 64 * cpr; i += SCAN_THREADS) {
                int r = i / cpr, k8 = i % cpr;
                g2l_16B(hread + (size_t)(rowg0 + r) * HH + k8 * 8,
                        stage + (size_t)r * Ktot + Kxp + k8 * 8);
            }
        }
        {
            const int cpr = Kxp / 8;
            for (int i = tid; i < 64 * cpr; i += SCAN_THREADS) {
                int r = i / cpr, k8 = i % cpr;
                g2l_16B(xin + ((size_t)(rowg0 + r) * TT + t) * Kxp + k8 * 8,
                        stage + (size_t)r * Ktot + k8 * 8);
            }
        }
        // gfx1250 prefetch of next step's x rows
        if (t + 1 < TT)
            __builtin_prefetch(xin + ((size_t)(rowg0 + (tid & 63)) * TT + (t + 1)) * Kxp, 0, 1);
        wait_async_done();
        __syncthreads();

        // ---- gates = bias + [x_t | h_{t-1}] x W^T   (4x WMMA per K step)
        v8f acc[4];
#pragma unroll
        for (int g = 0; g < 4; ++g) {
#pragma unroll
            for (int r = 0; r < 8; ++r) acc[g][r] = bv[g];
        }
        const __bf16* abase = stage + (size_t)arow * Ktot + koff;
        const int nk = (t == 0) ? nKKx : nKK;          // t=0: x-part only (h==0)
        for (int kk = 0; kk < nk; ++kk) {
            const __bf16* p = abase + kk * 32;
            v16bf a = *(const v16bf*)p;
#pragma unroll
            for (int g = 0; g < 4; ++g) {
                v16bf bf = *(const v16bf*)(wlds + (size_t)g * tile_halves +
                                           ((size_t)kk * 32 + lane) * 16);
                acc[g] = __builtin_amdgcn_wmma_f32_16x16x32_bf16(
                    false, a, false, bf, (short)0, acc[g], false, false);
            }
        }

        // ---- activations + cell update + h write (C/D layout: VGPR r ->
        //      M = r + 8*(lane>=16), N = lane&15)
#pragma unroll
        for (int r = 0; r < 8; ++r) {
            float iv = sigm(acc[0][r]);
            float fv = sigm(acc[1][r]);
            float gv = tanhf(acc[2][r]);
            float ov = sigm(acc[3][r]);
            float cv = fv * c[r] + iv * gv;
            c[r] = cv;
            float hv = ov * tanhf(cv);
            __bf16 hb = (__bf16)hv;
            int m = r + ((lane >> 4) << 3);
            int n = lane & 15;
            size_t grow = (size_t)(rowg0 + rbase + m);
            int gcol = hi * 16 + n;
            hwrite[grow * HH + gcol] = hb;
            hout[(grow * TT + t) * HH + gcol] = hb;
        }

        grid_barrier(counter, (unsigned)(NWG * (t + 1)));
    }
}

// ---------------------------------------------------------------------------
// FC head: out[b] = sigmoid( h_last[b,:] . fc_w + fc_b )
// ---------------------------------------------------------------------------
__global__ void fc_head_kernel(const __bf16* __restrict__ hout,   // layer-3 [B,T,H]
                               const float* __restrict__ fc_w,
                               const float* __restrict__ fc_b,
                               float* __restrict__ out) {
    int b = blockIdx.x * blockDim.x + threadIdx.x;
    if (b >= BB) return;
    const __bf16* hp = hout + ((size_t)b * TT + (TT - 1)) * HH;
    float s = fc_b[0];
    for (int j = 0; j < HH; ++j) s += (float)hp[j] * fc_w[j];
    out[b] = sigm(s);
}

// ---------------------------------------------------------------------------
extern "C" void kernel_launch(void* const* d_in, const int* in_sizes, int n_in,
                              void* d_out, int out_size, void* d_ws, size_t ws_size,
                              hipStream_t stream) {
    (void)in_sizes; (void)n_in; (void)out_size; (void)ws_size;
    const float* x      = (const float*)d_in[0];
    const float* w_ih0  = (const float*)d_in[1];
    const float* w_hh0  = (const float*)d_in[2];
    const float* b_ih0  = (const float*)d_in[3];
    const float* b_hh0  = (const float*)d_in[4];
    const float* w_ih_r = (const float*)d_in[5];   // [3, 2048, 512]
    const float* w_hh_r = (const float*)d_in[6];   // [3, 2048, 512]
    const float* b_ih_r = (const float*)d_in[7];   // [3, 2048]
    const float* b_hh_r = (const float*)d_in[8];   // [3, 2048]
    const float* fc_w   = (const float*)d_in[9];
    const float* fc_b   = (const float*)d_in[10];
    float* out = (float*)d_out;

    // --- workspace carve-up
    char* ws = (char*)d_ws;
    size_t off = 0;
    auto carve = [&](size_t bytes) -> char* {
        char* p = ws + off;
        off += (bytes + 255) & ~(size_t)255;
        return p;
    };
    unsigned* counter = (unsigned*)carve(256);
    __bf16* xbf   = (__bf16*)carve((size_t)BB * TT * 32 * 2);
    __bf16* hbuf0 = (__bf16*)carve((size_t)BB * TT * HH * 2);
    __bf16* hbuf1 = (__bf16*)carve((size_t)BB * TT * HH * 2);
    __bf16* hst0  = (__bf16*)carve((size_t)BB * HH * 2);
    __bf16* hst1  = (__bf16*)carve((size_t)BB * HH * 2);
    __bf16* wfrag[4];
    for (int l = 0; l < 4; ++l) wfrag[l] = (__bf16*)carve((size_t)GG * 1024 * 2);
    float* biasc[4];
    for (int l = 0; l < 4; ++l) biasc[l] = (float*)carve((size_t)GG * 4);

    // Allow >64KB dynamic LDS for the scan kernel (no-op if unsupported).
    (void)hipFuncSetAttribute((const void*)lstm_scan_kernel,
                              hipFuncAttributeMaxDynamicSharedMemorySize, 262144);

    // --- prep
    prep_x_kernel<<<1024, 256, 0, stream>>>(x, xbf);
    prep_wfrag_kernel<<<2048, 256, 0, stream>>>(w_ih0, w_hh0, II, 32, wfrag[0]);
    prep_bias_kernel<<<GG / 256, 256, 0, stream>>>(b_ih0, b_hh0, biasc[0]);
    for (int l = 1; l < 4; ++l) {
        prep_wfrag_kernel<<<2048, 256, 0, stream>>>(
            w_ih_r + (size_t)(l - 1) * GG * HH, w_hh_r + (size_t)(l - 1) * GG * HH,
            HH, HH, wfrag[l]);
        prep_bias_kernel<<<GG / 256, 256, 0, stream>>>(
            b_ih_r + (size_t)(l - 1) * GG, b_hh_r + (size_t)(l - 1) * GG, biasc[l]);
    }

    // --- 4 persistent recurrent scans
    const __bf16* lin[4]  = { xbf,  hbuf0, hbuf1, hbuf0 };
    __bf16*       lout[4] = { hbuf0, hbuf1, hbuf0, hbuf1 };
    const int     kxp[4]  = { 32, HH, HH, HH };
    for (int l = 0; l < 4; ++l) {
        const int Ktot = kxp[l] + HH;
        const size_t lds = ((size_t)4 * Ktot * 16 + (size_t)64 * Ktot) * 2;
        (void)hipMemsetAsync(counter, 0, sizeof(unsigned), stream);
        lstm_scan_kernel<<<NWG, SCAN_THREADS, lds, stream>>>(
            lin[l], kxp[l], wfrag[l], biasc[l], hst0, hst1, lout[l], counter);
    }

    // --- FC + sigmoid head
    fc_head_kernel<<<1, 256, 0, stream>>>(hbuf1, fc_w, fc_b, out);
}
